// _OvisDoubleAttn_65343632441851
// MI455X (gfx1250) — compile-verified, matrix operand order checked by reading, not run on hardware
//
#include <hip/hip_runtime.h>
#include <cstdint>
#include <cstddef>

typedef __attribute__((ext_vector_type(16))) __bf16 v16bf;
typedef __attribute__((ext_vector_type(8)))  __bf16 v8bf;
typedef __attribute__((ext_vector_type(8)))  float  v8f;

#define HID   2048
#define HEADS 16
#define HD    128
#define SEQ_I 2048
#define SEQ_T 512
#define SEQ_A 2560
#define BATCH 2

// ---------------------------------------------------------------------------
// WMMA fragment helpers (wave32; 16-bit A/B layout per CDNA5 ISA):
//   A frag (16x32, M x K):  lane L holds row (L&15); K offset +8 for L>=16.
//     v16bf elements [0..7]  = K  koff+0..7   (one b128 load)
//     v16bf elements [8..15] = K  koff+16..23 (one b128 load)
//   B frag (32x16, K x N) mirrors A with N playing the role of M, so loading
//   rows of W[N,K] (y = x.W^T) uses identical addressing.
// ---------------------------------------------------------------------------
__device__ __forceinline__ v16bf frag_cat(v8bf lo, v8bf hi) {
  return __builtin_shufflevector(lo, hi, 0,1,2,3,4,5,6,7,8,9,10,11,12,13,14,15);
}

__device__ __forceinline__ v16bf load_frag_p(const __bf16* p) {
  v8bf lo = *(const v8bf*)p;
  v8bf hi = *(const v8bf*)(p + 16);
  return frag_cat(lo, hi);
}

__device__ __forceinline__ v16bf load_frag(const __bf16* base, int ld, int row, int koff, int lane) {
  return load_frag_p(base + (size_t)(row + (lane & 15)) * ld + koff + ((lane >> 4) << 3));
}

__device__ __forceinline__ v8f wmma_bf16(v16bf a, v16bf b, v8f c) {
  // 8 args: (neg_a, A, neg_b, B, c_mod, C, reuse_a, reuse_b)
  return __builtin_amdgcn_wmma_f32_16x16x32_bf16(false, a, false, b, (short)0, c, false, false);
}

__device__ __forceinline__ float red_max16(float v) {
  v = fmaxf(v, __shfl_xor(v, 1, 32));
  v = fmaxf(v, __shfl_xor(v, 2, 32));
  v = fmaxf(v, __shfl_xor(v, 4, 32));
  v = fmaxf(v, __shfl_xor(v, 8, 32));
  return v;
}
__device__ __forceinline__ float red_sum16(float v) {
  v += __shfl_xor(v, 1, 32);
  v += __shfl_xor(v, 2, 32);
  v += __shfl_xor(v, 4, 32);
  v += __shfl_xor(v, 8, 32);
  return v;
}
__device__ __forceinline__ float red_sum32(float v) {
  v = red_sum16(v);
  v += __shfl_xor(v, 16, 32);
  return v;
}

// ---------------------------------------------------------------------------
// fp32 -> bf16 cast
// ---------------------------------------------------------------------------
__global__ void cast_f32_bf16(const float* __restrict__ in, __bf16* __restrict__ out, int n) {
  int i = blockIdx.x * blockDim.x + threadIdx.x;
  if (i < n) out[i] = (__bf16)in[i];
}

// ---------------------------------------------------------------------------
// GEMM: C[M,N] = A[M,K](bf16) . W[N,K]^T(bf16) + bias[N], fp32 out.
// block = 256 (8 waves). Wave w -> rows [blockIdx.y*128 + 16w, +16), cols
// [blockIdx.x*64, +64). Software-pipelined: fragments for k-step i+1 are
// loaded before the WMMAs of step i issue, so global latency overlaps the
// matrix pipe instead of serializing behind s_wait_loadcnt 0.
// ---------------------------------------------------------------------------
__global__ void gemm_bf16_wmma(const __bf16* __restrict__ A, const __bf16* __restrict__ W,
                               const float* __restrict__ bias, float* __restrict__ C,
                               int M, int N, int K) {
  const int lane  = threadIdx.x & 31;
  const int wave  = threadIdx.x >> 5;
  const int row0  = blockIdx.y * 128 + wave * 16;
  const int n0    = blockIdx.x * 64;
  const int half8 = (lane >> 4) << 3;   // K offset (+8) for hi half; also C-row (+8)
  const int l15   = lane & 15;

  const __bf16* pa  = A + (size_t)(row0 + l15) * K + half8;
  const __bf16* pw0 = W + (size_t)(n0 +  0 + l15) * K + half8;
  const __bf16* pw1 = W + (size_t)(n0 + 16 + l15) * K + half8;
  const __bf16* pw2 = W + (size_t)(n0 + 32 + l15) * K + half8;
  const __bf16* pw3 = W + (size_t)(n0 + 48 + l15) * K + half8;

  v8f acc0 = {}, acc1 = {}, acc2 = {}, acc3 = {};

  // prologue: fragments for k-step 0
  v16bf af = load_frag_p(pa);
  v16bf b0 = load_frag_p(pw0);
  v16bf b1 = load_frag_p(pw1);
  v16bf b2 = load_frag_p(pw2);
  v16bf b3 = load_frag_p(pw3);

  for (int k0 = 32; k0 < K; k0 += 32) {
    // issue next-step loads first (independent of current WMMAs)
    v16bf afn = load_frag_p(pa  + k0);
    v16bf b0n = load_frag_p(pw0 + k0);
    v16bf b1n = load_frag_p(pw1 + k0);
    v16bf b2n = load_frag_p(pw2 + k0);
    v16bf b3n = load_frag_p(pw3 + k0);
    if (k0 + 256 <= K) {               // global_prefetch_b8: warm L2 ~7 tiles ahead
      __builtin_prefetch(pa  + k0 + 224, 0, 1);
      __builtin_prefetch(pw0 + k0 + 224, 0, 1);
      __builtin_prefetch(pw2 + k0 + 224, 0, 1);
    }
    acc0 = wmma_bf16(af, b0, acc0);
    acc1 = wmma_bf16(af, b1, acc1);
    acc2 = wmma_bf16(af, b2, acc2);
    acc3 = wmma_bf16(af, b3, acc3);
    af = afn; b0 = b0n; b1 = b1n; b2 = b2n; b3 = b3n;
  }
  // epilogue k-step
  acc0 = wmma_bf16(af, b0, acc0);
  acc1 = wmma_bf16(af, b1, acc1);
  acc2 = wmma_bf16(af, b2, acc2);
  acc3 = wmma_bf16(af, b3, acc3);

  #pragma unroll
  for (int r = 0; r < 8; ++r) {
    const int m = row0 + r + half8;
    const int n = n0 + l15;
    float* c = C + (size_t)m * N + n;
    c[ 0] = acc0[r] + bias[n];
    c[16] = acc1[r] + bias[n + 16];
    c[32] = acc2[r] + bias[n + 32];
    c[48] = acc3[r] + bias[n + 48];
  }
}

// ---------------------------------------------------------------------------
// Post-QKV: per (b,s,h) row of 128: RMS-norm(q,k) -> RoPE(q,k) -> fold the
// 1/sqrt(D) softmax scale into q -> pack bf16 into [B,H,SEQ_A,D]. v is packed
// unchanged. One wave per (b,s,h); lane owns 4 consecutive d (2 rope pairs).
// ---------------------------------------------------------------------------
__global__ void postqkv_kernel(const float* __restrict__ qraw, const float* __restrict__ kraw,
                               const float* __restrict__ vraw,
                               const float* __restrict__ cosT, const float* __restrict__ sinT,
                               const float* __restrict__ nqw,  const float* __restrict__ nkw,
                               __bf16* __restrict__ qb, __bf16* __restrict__ kb, __bf16* __restrict__ vb,
                               int seq, int s_off, float qscale) {
  const int lane = threadIdx.x & 31;
  const int wid  = blockIdx.x * 8 + (threadIdx.x >> 5);
  const int h = wid & 15;
  const int s = (wid >> 4) % seq;
  const int b = wid / (16 * seq);
  const int d0 = lane * 4;

  const size_t inoff  = ((size_t)(b * seq + s)) * HID + h * HD + d0;
  const size_t outoff = (((size_t)(b * HEADS + h)) * SEQ_A + (s_off + s)) * HD + d0;

  float4 qv = *(const float4*)(qraw + inoff);
  float4 kv = *(const float4*)(kraw + inoff);
  float4 vv = *(const float4*)(vraw + inoff);

  float qs = qv.x * qv.x + qv.y * qv.y + qv.z * qv.z + qv.w * qv.w;
  float ks = kv.x * kv.x + kv.y * kv.y + kv.z * kv.z + kv.w * kv.w;
  qs = red_sum32(qs);
  ks = red_sum32(ks);
  const float qr = rsqrtf(qs * (1.0f / 128.0f) + 1e-6f);
  const float kr = rsqrtf(ks * (1.0f / 128.0f) + 1e-6f);

  const float qn0 = qv.x * qr * nqw[d0], qn1 = qv.y * qr * nqw[d0 + 1];
  const float qn2 = qv.z * qr * nqw[d0 + 2], qn3 = qv.w * qr * nqw[d0 + 3];
  const float kn0 = kv.x * kr * nkw[d0], kn1 = kv.y * kr * nkw[d0 + 1];
  const float kn2 = kv.z * kr * nkw[d0 + 2], kn3 = kv.w * kr * nkw[d0 + 3];

  const float* cp = cosT + (size_t)s * HD + d0;
  const float* sp = sinT + (size_t)s * HD + d0;
  const float c0 = cp[0], c1 = cp[1], c2 = cp[2], c3 = cp[3];
  const float n0 = sp[0], n1 = sp[1], n2 = sp[2], n3 = sp[3];

  // rot of (x0,x1,x2,x3) = (-x1, x0, -x3, x2); out = x*cos + rot*sin
  qb[outoff + 0] = (__bf16)((qn0 * c0 - qn1 * n0) * qscale);
  qb[outoff + 1] = (__bf16)((qn1 * c1 + qn0 * n1) * qscale);
  qb[outoff + 2] = (__bf16)((qn2 * c2 - qn3 * n2) * qscale);
  qb[outoff + 3] = (__bf16)((qn3 * c3 + qn2 * n3) * qscale);
  kb[outoff + 0] = (__bf16)(kn0 * c0 - kn1 * n0);
  kb[outoff + 1] = (__bf16)(kn1 * c1 + kn0 * n1);
  kb[outoff + 2] = (__bf16)(kn2 * c2 - kn3 * n2);
  kb[outoff + 3] = (__bf16)(kn3 * c3 + kn2 * n3);
  vb[outoff + 0] = (__bf16)vv.x;
  vb[outoff + 1] = (__bf16)vv.y;
  vb[outoff + 2] = (__bf16)vv.z;
  vb[outoff + 3] = (__bf16)vv.w;
}

// ---------------------------------------------------------------------------
// Flash attention over S=2560, D=128. Block = 256 (8 waves) handles one
// (b,h, 128 q-rows). Double-buffered K/V LDS tiles: global loads for tile i+1
// issue before the 16 WMMAs of tile i, LDS stores for i+1 follow compute,
// one barrier per iteration. P converted C-layout -> A-layout via per-wave
// LDS buffer (same-wave DS ops are in-order).
// ---------------------------------------------------------------------------
__global__ void attn_kernel(const __bf16* __restrict__ Q, const __bf16* __restrict__ Km,
                            const __bf16* __restrict__ V,
                            __bf16* __restrict__ img_o, __bf16* __restrict__ txt_o) {
  __shared__ __bf16 Kt[2][32 * 128];    // [buf][kcol][d]
  __shared__ __bf16 Vt[2][128 * 32];    // [buf][d][kcol]  (transposed)
  __shared__ __bf16 Pt[8][16 * 32];     // per-wave P tile [qrow][kcol]

  const int tid   = threadIdx.x;
  const int lane  = tid & 31;
  const int wave  = tid >> 5;
  const int bh    = blockIdx.y;
  const int b     = bh >> 4;
  const int h     = bh & 15;
  const int half8 = (lane >> 4) << 3;
  const int l15   = lane & 15;

  const size_t base = (size_t)bh * SEQ_A * HD;
  const __bf16* q = Q  + base;
  const __bf16* k = Km + base;
  const __bf16* v = V  + base;

  const int qrow0 = blockIdx.x * 128 + wave * 16;

  // cooperative-stage geometry: thread covers elems [tid*8, +8) and [+2048, +8)
  const int r0 = (tid * 8) >> 7,          c0 = (tid * 8) & 127;
  const int r1 = ((tid + 256) * 8) >> 7,  c1 = ((tid + 256) * 8) & 127;

  v16bf qf[4];
  #pragma unroll
  for (int t = 0; t < 4; ++t) qf[t] = load_frag(q, HD, qrow0, t * 32, lane);

  v8f o[8];
  #pragma unroll
  for (int dt = 0; dt < 8; ++dt) o[dt] = (v8f){};
  float mrow[8], lrow[8];
  #pragma unroll
  for (int r = 0; r < 8; ++r) { mrow[r] = -1e30f; lrow[r] = 0.0f; }

  // prologue: stage tile 0 into buffer 0
  v8bf ka0 = *(const v8bf*)(k + (size_t)r0 * HD + c0);
  v8bf ka1 = *(const v8bf*)(k + (size_t)r1 * HD + c1);
  v8bf va0 = *(const v8bf*)(v + (size_t)r0 * HD + c0);
  v8bf va1 = *(const v8bf*)(v + (size_t)r1 * HD + c1);
  *(v8bf*)(&Kt[0][r0 * 128 + c0]) = ka0;
  *(v8bf*)(&Kt[0][r1 * 128 + c1]) = ka1;
  #pragma unroll
  for (int j = 0; j < 8; ++j) Vt[0][(c0 + j) * 32 + r0] = va0[j];
  #pragma unroll
  for (int j = 0; j < 8; ++j) Vt[0][(c1 + j) * 32 + r1] = va1[j];
  __syncthreads();

  for (int kp = 0; kp < SEQ_A; kp += 32) {
    const int  cur  = (kp >> 5) & 1;
    const bool more = (kp + 32) < SEQ_A;

    // issue next tile's global loads before compute (overlap HBM with WMMA)
    if (more) {
      const __bf16* kn = k + (size_t)(kp + 32) * HD;
      const __bf16* vn = v + (size_t)(kp + 32) * HD;
      ka0 = *(const v8bf*)(kn + (size_t)r0 * HD + c0);
      ka1 = *(const v8bf*)(kn + (size_t)r1 * HD + c1);
      va0 = *(const v8bf*)(vn + (size_t)r0 * HD + c0);
      va1 = *(const v8bf*)(vn + (size_t)r1 * HD + c1);
    }

    // scores: S[16q x 32k] = Q . K^T  (two 16x16 accumulators)
    v8f s0 = {}, s1 = {};
    #pragma unroll
    for (int t = 0; t < 4; ++t) {
      const __bf16* kb0 = &Kt[cur][(size_t)l15 * 128 + t * 32 + half8];
      const __bf16* kb1 = &Kt[cur][(size_t)(16 + l15) * 128 + t * 32 + half8];
      v16bf bf0 = load_frag_p(kb0);
      v16bf bf1 = load_frag_p(kb1);
      s0 = wmma_bf16(qf[t], bf0, s0);
      s1 = wmma_bf16(qf[t], bf1, s1);
    }

    // online softmax (softmax scale already folded into q)
    #pragma unroll
    for (int r = 0; r < 8; ++r) {
      float smax = red_max16(fmaxf(s0[r], s1[r]));
      const float nm = fmaxf(mrow[r], smax);
      const float f  = __expf(mrow[r] - nm);
      mrow[r] = nm;
      const float p0 = __expf(s0[r] - nm);
      const float p1 = __expf(s1[r] - nm);
      lrow[r] = lrow[r] * f + red_sum16(p0 + p1);
      #pragma unroll
      for (int dt = 0; dt < 8; ++dt) o[dt][r] = o[dt][r] * f;
      const int prow = r + half8;
      Pt[wave][prow * 32 + l15]      = (__bf16)p0;
      Pt[wave][prow * 32 + 16 + l15] = (__bf16)p1;
    }

    // reload P as A fragment (same-wave DS ops are in-order)
    v16bf pf = load_frag_p(&Pt[wave][l15 * 32 + half8]);

    // O += P . V over 8 d-tiles
    #pragma unroll
    for (int dt = 0; dt < 8; ++dt) {
      v16bf vf = load_frag_p(&Vt[cur][(size_t)(dt * 16 + l15) * 32 + half8]);
      o[dt] = wmma_bf16(pf, vf, o[dt]);
    }

    // stage next tile into the other buffer, single barrier per iteration
    if (more) {
      const int nxt = cur ^ 1;
      *(v8bf*)(&Kt[nxt][r0 * 128 + c0]) = ka0;
      *(v8bf*)(&Kt[nxt][r1 * 128 + c1]) = ka1;
      #pragma unroll
      for (int j = 0; j < 8; ++j) Vt[nxt][(c0 + j) * 32 + r0] = va0[j];
      #pragma unroll
      for (int j = 0; j < 8; ++j) Vt[nxt][(c1 + j) * 32 + r1] = va1[j];
    }
    __syncthreads();
  }

  // epilogue: O/l -> bf16 into [B, seq_stream, HID]
  #pragma unroll
  for (int r = 0; r < 8; ++r) {
    const float inv = 1.0f / lrow[r];
    const int qg = qrow0 + r + half8;
    #pragma unroll
    for (int dt = 0; dt < 8; ++dt) {
      const float val = o[dt][r] * inv;
      const int d = dt * 16 + l15;
      if (qg < SEQ_I)
        img_o[((size_t)b * SEQ_I + qg) * HID + h * HD + d] = (__bf16)val;
      else
        txt_o[((size_t)b * SEQ_T + (qg - SEQ_I)) * HID + h * HD + d] = (__bf16)val;
    }
  }
}

// ---------------------------------------------------------------------------
// Host side
// ---------------------------------------------------------------------------
extern "C" void kernel_launch(void* const* d_in, const int* in_sizes, int n_in,
                              void* d_out, int out_size, void* d_ws, size_t ws_size,
                              hipStream_t stream) {
  (void)in_sizes; (void)n_in; (void)out_size; (void)ws_size;

  const float* img      = (const float*)d_in[0];
  const float* txt      = (const float*)d_in[1];
  const float* img_cos  = (const float*)d_in[2];
  const float* img_sin  = (const float*)d_in[3];
  const float* txt_cos  = (const float*)d_in[4];
  const float* txt_sin  = (const float*)d_in[5];
  const float* w_q      = (const float*)d_in[6];
  const float* b_q      = (const float*)d_in[7];
  const float* w_k      = (const float*)d_in[8];
  const float* b_k      = (const float*)d_in[9];
  const float* w_v      = (const float*)d_in[10];
  const float* b_v      = (const float*)d_in[11];
  const float* w_out    = (const float*)d_in[12];
  const float* b_out    = (const float*)d_in[13];
  const float* w_aq     = (const float*)d_in[14];
  const float* b_aq     = (const float*)d_in[15];
  const float* w_ak     = (const float*)d_in[16];
  const float* b_ak     = (const float*)d_in[17];
  const float* w_av     = (const float*)d_in[18];
  const float* b_av     = (const float*)d_in[19];
  const float* w_aout   = (const float*)d_in[20];
  const float* b_aout   = (const float*)d_in[21];
  const float* norm_q_w = (const float*)d_in[22];
  const float* norm_k_w = (const float*)d_in[23];
  const float* norm_aq_w= (const float*)d_in[24];
  const float* norm_ak_w= (const float*)d_in[25];

  const size_t WN  = (size_t)HID * HID;          // weight elems
  const size_t MI  = (size_t)BATCH * SEQ_I;      // 4096 img rows
  const size_t MT  = (size_t)BATCH * SEQ_T;      // 1024 txt rows
  const size_t XI  = MI * HID;
  const size_t XT  = MT * HID;
  const size_t QKV = (size_t)BATCH * HEADS * SEQ_A * HD;

  char* wsp = (char*)d_ws;
  size_t off = 0;
  auto alloc_bf = [&](size_t n) { __bf16* p = (__bf16*)(wsp + off); off += ((n * 2 + 255) & ~(size_t)255); return p; };
  auto alloc_f  = [&](size_t n) { float*  p = (float*) (wsp + off); off += ((n * 4 + 255) & ~(size_t)255); return p; };

  __bf16* wbq   = alloc_bf(WN);  __bf16* wbk   = alloc_bf(WN);  __bf16* wbv   = alloc_bf(WN);
  __bf16* wbo   = alloc_bf(WN);  __bf16* wbaq  = alloc_bf(WN);  __bf16* wbak  = alloc_bf(WN);
  __bf16* wbav  = alloc_bf(WN);  __bf16* wbao  = alloc_bf(WN);
  __bf16* xbi   = alloc_bf(XI);  __bf16* xbt   = alloc_bf(XT);
  float*  iq    = alloc_f(XI);   float*  ik    = alloc_f(XI);   float*  iv    = alloc_f(XI);
  float*  tq    = alloc_f(XT);   float*  tk    = alloc_f(XT);   float*  tv    = alloc_f(XT);
  __bf16* qb    = alloc_bf(QKV); __bf16* kb    = alloc_bf(QKV); __bf16* vb    = alloc_bf(QKV);
  __bf16* iatt  = alloc_bf(XI);  __bf16* tatt  = alloc_bf(XT);

  auto cast = [&](const float* src, __bf16* dst, size_t n) {
    cast_f32_bf16<<<(unsigned)((n + 255) / 256), 256, 0, stream>>>(src, dst, (int)n);
  };
  cast(w_q, wbq, WN);   cast(w_k, wbk, WN);   cast(w_v, wbv, WN);   cast(w_out, wbo, WN);
  cast(w_aq, wbaq, WN); cast(w_ak, wbak, WN); cast(w_av, wbav, WN); cast(w_aout, wbao, WN);
  cast(img, xbi, XI);   cast(txt, xbt, XT);

  auto gemm = [&](const __bf16* A, const __bf16* W, const float* bias, float* C, int M) {
    dim3 grid(HID / 64, M / 128);
    gemm_bf16_wmma<<<grid, 256, 0, stream>>>(A, W, bias, C, M, HID, HID);
  };
  // QKV projections
  gemm(xbi, wbq,  b_q,  iq, (int)MI);
  gemm(xbi, wbk,  b_k,  ik, (int)MI);
  gemm(xbi, wbv,  b_v,  iv, (int)MI);
  gemm(xbt, wbaq, b_aq, tq, (int)MT);
  gemm(xbt, wbak, b_ak, tk, (int)MT);
  gemm(xbt, wbav, b_av, tv, (int)MT);

  // RMS-norm + RoPE + pack (softmax scale folded into q)
  const float qscale = 0.08838834764831845f;   // 1/sqrt(128)
  postqkv_kernel<<<(unsigned)(BATCH * SEQ_I * HEADS / 8), 256, 0, stream>>>(
      iq, ik, iv, img_cos, img_sin, norm_q_w, norm_k_w, qb, kb, vb, SEQ_I, 0, qscale);
  postqkv_kernel<<<(unsigned)(BATCH * SEQ_T * HEADS / 8), 256, 0, stream>>>(
      tq, tk, tv, txt_cos, txt_sin, norm_aq_w, norm_ak_w, qb, kb, vb, SEQ_T, SEQ_I, qscale);

  // flash attention
  attn_kernel<<<dim3(SEQ_A / 128, BATCH * HEADS), 256, 0, stream>>>(qb, kb, vb, iatt, tatt);

  // output projections straight into d_out (img_out flat, then txt_out)
  float* out_img = (float*)d_out;
  float* out_txt = out_img + XI;
  gemm(iatt, wbo,  b_out,  out_img, (int)MI);
  gemm(tatt, wbao, b_aout, out_txt, (int)MT);
}
